// BasicBlock_case1_45835890983453
// MI455X (gfx1250) — 8-bit matrix operand order repaired and checked by reading, compiled, not run on hardware
//
#include <hip/hip_runtime.h>

typedef int v8i __attribute__((ext_vector_type(8)));
typedef int v4i __attribute__((ext_vector_type(4)));
typedef __attribute__((address_space(1))) v4i gv4i;
typedef __attribute__((address_space(3))) v4i lv4i;

#if __has_builtin(__builtin_amdgcn_global_load_async_to_lds_b128) && \
    __has_builtin(__builtin_amdgcn_s_wait_asynccnt)
#define HAVE_ASYNC_LDS 1
#endif

#define NB    64
#define C_CH  256
#define HH    28
#define WW    28
#define HWSZ  784
#define HP    30
#define WP    30
#define KDIM  2304
#define XB_OFF   0u
#define WB_OFF   14745600u
#define ST_OFF   15335424u
#define Y_OFF    15337472u

__global__ void zero_stats_kernel(float* __restrict__ stats) {
    int i = threadIdx.x;
    if (i < 2 * C_CH) stats[i] = 0.0f;
}

__global__ void bin_x_kernel(const float* __restrict__ x, signed char* __restrict__ xb) {
    int e = blockIdx.x * blockDim.x + threadIdx.x;
    int ci = e & 255;
    int p  = e >> 8;
    int wp = p % WP; p /= WP;
    int hp = p % HP;
    int np = p / HP;
    signed char v = 0;
    if (hp >= 1 && hp <= HH && wp >= 1 && wp <= WW) {
        float xv = x[((np * C_CH + ci) * HH + (hp - 1)) * WW + (wp - 1)];
        v = (xv > 0.0f) ? 1 : ((xv < 0.0f) ? -1 : 0);
    }
    xb[e] = v;
}

__global__ void bin_w_kernel(const float* __restrict__ w, signed char* __restrict__ wb) {
    int e = blockIdx.x * blockDim.x + threadIdx.x;
    int ci = e & 255;
    int p  = e >> 8;
    int kt = p % 9;
    int co = p / 9;

    int j   = ci & 63;
    int ci0 = ci & ~63;
    int h = j >> 5, r = (j >> 2) & 7, b = j & 3;
    int k = 32 * (r >> 2) + 16 * h + 4 * (r & 3) + b;
    int d   = (k >> 2) & 15;
    int rp  = 2 * (d >> 2) + (d & 1);
    int hp2 = (d >> 1) & 1;
    int src = j;
    int srcci = ci0 + src;

    float wv = w[(co * C_CH + srcci) * 9 + kt];
    wb[(size_t)co * KDIM + kt * C_CH + ci] = (wv > 0.0f) ? 1 : ((wv < 0.0f) ? -1 : 0);
}

__device__ __forceinline__ v8i frag32(const signed char* p) {
    const int4* q = (const int4*)p;
    int4 lo = q[0], hi = q[1];
    v8i f;
    f[0] = lo.x; f[1] = lo.y; f[2] = lo.z; f[3] = lo.w;
    f[4] = hi.x; f[5] = hi.y; f[6] = hi.z; f[7] = hi.w;
    return f;
}

__global__ __launch_bounds__(256) void conv_wmma_kernel(
    const signed char* __restrict__ xb, const signed char* __restrict__ wb,
    const float* __restrict__ scale, float* __restrict__ yws,
    float* __restrict__ stats) {

    __shared__ signed char bsm[16 * KDIM];

    const int tid    = threadIdx.x;
    const int lane   = tid & 31;
    const int waveid = tid >> 5;
    const int l15    = lane & 15;
    const int khalf  = lane >> 4;

    const int blk = blockIdx.x;
    const int nt  = blk & 15;
    const int q   = ((blk >> 4) << 3) + waveid;
    const int mt0 = q << 1;
    const int mt1 = mt0 + 1;

    {
        const int co_l = tid >> 4;
        const int sub  = tid & 15;
        const int soff = (sub * 16) ^ ((co_l & 7) << 5);
        const signed char* gsrc = wb + (size_t)(nt * 16 + co_l) * KDIM + sub * 16;
        signed char* ldst = &bsm[co_l * KDIM + soff];
#ifdef HAVE_ASYNC_LDS
#pragma unroll
        for (int jj = 0; jj < 9; ++jj) {
            __builtin_amdgcn_global_load_async_to_lds_b128(
                (gv4i*)(gsrc + jj * 256), (lv4i*)(ldst + jj * 256), 0, 0);
        }
        __builtin_amdgcn_s_wait_asynccnt(0);
#else
#pragma unroll
        for (int jj = 0; jj < 9; ++jj)
            *(int4*)(ldst + jj * 256) = *(const int4*)(gsrc + jj * 256);
#endif
    }
    __syncthreads();

    const int m0 = mt0 * 16 + l15;
    int t = m0;
    const int w0   = t % WW; t /= WW;
    const int h0   = t % HH;
    const int n0   = t / HH;
    const int m1   = m0 + 16;
    t = m1;
    const int w1   = t % WW; t /= WW;
    const int h1   = t % HH;
    const int n1   = t / HH;

    const signed char* aBase0 = xb + (((size_t)((n0 * HP + h0) * WP + w0)) << 8) + (khalf << 5);
    const signed char* aBase1 = xb + (((size_t)((n1 * HP + h1) * WP + w1)) << 8) + (khalf << 5);
    const signed char* bBase  = &bsm[l15 * KDIM];
    const int swzB = (l15 & 7) << 5;

    v8i acc0 = {0, 0, 0, 0, 0, 0, 0, 0};
    v8i acc1 = {0, 0, 0, 0, 0, 0, 0, 0};

#pragma unroll 4
    for (int kk = 0; kk < 36; ++kk) {
        const int kt  = kk >> 2;
        const int kh  = kt / 3;
        const int kw  = kt - kh * 3;
        const int ci0 = (kk & 3) << 6;
        const int aoff = ((kh * WP + kw) << 8) + ci0;
        const int boff = ((kt << 8) + ci0 + (khalf << 5)) ^ swzB;

        if ((kk & 3) == 0 && kt < 8) {
            const int kt2 = kt + 1;
            const int kh2 = kt2 / 3, kw2 = kt2 - kh2 * 3;
            __builtin_prefetch(aBase0 + ((kh2 * WP + kw2) << 8), 0, 0);
        }

        v8i B  = frag32(bBase + boff);
        v8i A0 = frag32(aBase0 + aoff);
        v8i A1 = frag32(aBase1 + aoff);

        acc0 = __builtin_amdgcn_wmma_i32_16x16x64_iu8(true, A0, true, B, acc0, false, false);
        acc1 = __builtin_amdgcn_wmma_i32_16x16x64_iu8(true, A1, true, B, acc1, false, false);
    }

    const int co   = nt * 16 + l15;
    const float sc = scale[co];
    float s = 0.0f, ss = 0.0f;
#pragma unroll
    for (int r = 0; r < 8; ++r) {
        const int mg = mt0 * 16 + (khalf << 3) + r;
        int t2 = mg;
        const int w2 = t2 % WW; t2 /= WW;
        const int h2 = t2 % HH;
        const int n2 = t2 / HH;
        const float y = sc * (float)acc0[r];
        yws[((size_t)(n2 * C_CH + co)) * HWSZ + h2 * WW + w2] = y;
        s += y;  ss += y * y;
    }
#pragma unroll
    for (int r = 0; r < 8; ++r) {
        const int mg = mt1 * 16 + (khalf << 3) + r;
        int t2 = mg;
        const int w2 = t2 % WW; t2 /= WW;
        const int h2 = t2 % HH;
        const int n2 = t2 / HH;
        const float y = sc * (float)acc1[r];
        yws[((size_t)(n2 * C_CH + co)) * HWSZ + h2 * WW + w2] = y;
        s += y;  ss += y * y;
    }
    s  += __shfl_xor(s, 16);
    ss += __shfl_xor(ss, 16);
    if (khalf == 0) {
        atomicAdd(&stats[co], s);
        atomicAdd(&stats[C_CH + co], ss);
    }
}

__global__ void bn_res_kernel(const float* __restrict__ yws, const float* __restrict__ x,
                              const float* __restrict__ stats,
                              const float* __restrict__ gamma, const float* __restrict__ beta,
                              float* __restrict__ out) {
    size_t i = (size_t)blockIdx.x * blockDim.x + threadIdx.x;
    const int co = (int)((i / HWSZ) % C_CH);
    const float inv = 1.0f / (float)(NB * HWSZ);
    const float mu  = stats[co] * inv;
    const float var = stats[C_CH + co] * inv - mu * mu;
    const float y   = yws[i];
    out[i] = gamma[co] * (y - mu) * rsqrtf(var + 1e-5f) + beta[co] + x[i];
}

extern "C" void kernel_launch(void* const* d_in, const int* in_sizes, int n_in,
                              void* d_out, int out_size, void* d_ws, size_t ws_size,
                              hipStream_t stream) {
    const float* x     = (const float*)d_in[0];
    const float* w     = (const float*)d_in[1];
    const float* scale = (const float*)d_in[2];
    const float* gamma = (const float*)d_in[3];
    const float* beta  = (const float*)d_in[4];
    float* out = (float*)d_out;

    char* ws = (char*)d_ws;
    signed char* xb = (signed char*)(ws + XB_OFF);
    signed char* wb = (signed char*)(ws + WB_OFF);
    float* stats    = (float*)(ws + ST_OFF);
    float* yws      = (float*)(ws + Y_OFF);

    zero_stats_kernel<<<1, 512, 0, stream>>>(stats);
    bin_x_kernel<<<57600, 256, 0, stream>>>(x, xb);
    bin_w_kernel<<<2304, 256, 0, stream>>>(w, wb);
    conv_wmma_kernel<<<3136, 256, 0, stream>>>(xb, wb, scale, yws, stats);
    bn_res_kernel<<<50176, 256, 0, stream>>>(yws, x, stats, gamma, beta, out);
}
